// SparseAutoencoder_87806311400057
// MI455X (gfx1250) — matrix operand order repaired and checked by reading, not compiled, not run on hardware
//
#include <hip/hip_runtime.h>
#include <hip/hip_bf16.h>

// ---------------- problem constants ----------------
#define B_DIM   4096
#define D_DIM   768
#define M_DIM   16384
#define K_SEL   32
#define MK_SEL  128
#define AUX_SEL 64
#define THRESH  256
#define NEG_FLT_MAX (-3.402823466e+38f)

typedef __bf16 bf16_t;
typedef bf16_t v16bf __attribute__((ext_vector_type(16)));
typedef float  v8f   __attribute__((ext_vector_type(8)));
typedef unsigned int v4u __attribute__((ext_vector_type(4)));
typedef int v4i __attribute__((ext_vector_type(4)));
typedef int v8i __attribute__((ext_vector_type(8)));

#if __has_builtin(__builtin_amdgcn_tensor_load_to_lds)
#define HAVE_TDM 1
#endif

// workspace layout (bytes)
#define OFF_XC   ((size_t)0)                              // B*D bf16   = 6 MB
#define OFF_WE   ((size_t)(B_DIM * D_DIM * 2))            // M*D bf16   = 24 MB
#define OFF_PRE  (OFF_WE + (size_t)M_DIM * D_DIM * 2)     // B*M f32    = 256 MB
#define OFF_SELV (OFF_PRE + (size_t)B_DIM * M_DIM * 4)    // B*128 f32
#define OFF_SELI (OFF_SELV + (size_t)B_DIM * MK_SEL * 4)  // B*128 i32

// ---------------------------------------------------------------------------
// Kernel 1: convert x - input_bias and W_enc to bf16
// ---------------------------------------------------------------------------
__global__ __launch_bounds__(256) void sae_convert(
    const float* __restrict__ x, const float* __restrict__ W_enc,
    const float* __restrict__ ib, bf16_t* __restrict__ xcb,
    bf16_t* __restrict__ web) {
  size_t i = (size_t)blockIdx.x * 256 + threadIdx.x;
  const size_t NX = (size_t)B_DIM * D_DIM;
  const size_t NW = (size_t)M_DIM * D_DIM;
  if (i < NX) {
    size_t d = i - (i / D_DIM) * D_DIM;
    xcb[i] = (bf16_t)(x[i] - ib[d]);
  }
  if (i < NW) web[i] = (bf16_t)W_enc[i];
}

// ---------------------------------------------------------------------------
// Kernel 2: pre_act[b,m] = xc[b,:] . W_enc[m,:] + neuron_bias[m]   (bf16 WMMA)
// WG = 8 waves; WG tile = 32 rows (b) x 256 cols (m).
// Each wave: 2x2 register-blocked 16x16 tiles (32b x 32m, 4 accumulators) so
// each k-step is 2 A-frag + 2 B-frag loads feeding 4 WMMAs (1 load/WMMA).
// A-panel (32x768 bf16 = 48 KB) staged in LDS via the Tensor Data Mover.
// ---------------------------------------------------------------------------
static __device__ __forceinline__ v16bf ld2x16(const bf16_t* p0, const bf16_t* p1) {
  union { uint4 u[2]; v16bf v; } t;
  t.u[0] = *(const uint4*)p0;
  t.u[1] = *(const uint4*)p1;
  return t.v;
}

__global__ __launch_bounds__(256) void sae_gemm(
    const bf16_t* __restrict__ xc, const bf16_t* __restrict__ we,
    const float* __restrict__ nb, float* __restrict__ pre) {
  __shared__ __align__(16) bf16_t sA[32 * D_DIM];  // 48 KB A-panel

  const int t = threadIdx.x;
  const int b0 = blockIdx.y * 32;
  const int m_blk = blockIdx.x * 256;

#ifdef HAVE_TDM
  // ---- TDM: async 2D tile load (768 x 32 rows of bf16) global -> LDS ----
  if (t < 32) {
    const unsigned lds_off = (unsigned)(size_t)(void*)&sA[0];
    const unsigned long long ga =
        (unsigned long long)(size_t)(const void*)(xc + (size_t)b0 * D_DIM);
    // D# group0: count=1 | lds_addr | global_addr[56:0] | type=2
    v4u g0 = { 1u, lds_off, (unsigned)ga,
               (unsigned)((ga >> 32) & 0x01FFFFFFu) | 0x80000000u };
    // D# group1: data_size=2B(enc 1); tensor_dim0=768; tensor_dim1=4096;
    //            tile_dim0=768; tile_dim1=32; dim0_stride=768
    v8i g1 = { (int)(1u << 16),
               (int)((768u & 0xFFFFu) << 16),
               (int)(((768u >> 16) & 0xFFFFu) | ((4096u & 0xFFFFu) << 16)),
               (int)(((4096u >> 16) & 0xFFFFu) | ((768u & 0xFFFFu) << 16)),
               (int)32u,
               (int)768,
               0, 0 };
    v4i g2 = { 0, 0, 0, 0 };
    v4i g3 = { 0, 0, 0, 0 };
#if __clang_major__ >= 23
    v8i g4 = { 0, 0, 0, 0, 0, 0, 0, 0 };
    __builtin_amdgcn_tensor_load_to_lds(g0, g1, g2, g3, g4, 0);
#else
    __builtin_amdgcn_tensor_load_to_lds(g0, g1, g2, g3, 0);
#endif
    __builtin_amdgcn_s_wait_tensorcnt(0);
  }
#else
  {
    const uint* src = (const uint*)(xc + (size_t)b0 * D_DIM);
    uint* dst = (uint*)sA;
    for (int i = t; i < 32 * D_DIM / 2; i += 256) dst[i] = src[i];
  }
#endif
  __syncthreads();

  const int wv = t >> 5, lane = t & 31;
  const int nIdx = lane & 15, half = lane >> 4;
  const int m0 = m_blk + wv * 32 + nIdx;       // m-subtile 0
  const int m1 = m0 + 16;                      // m-subtile 1

  // A fragments: lane row = lane%16 (+16 for b-subtile 1); chunks half*8, +16
  const bf16_t* aptr0 = sA + nIdx * D_DIM + half * 8;
  const bf16_t* aptr1 = aptr0 + 16 * D_DIM;
  // B fragments: chunks half*8, +16 (same as A)
  const bf16_t* bptr0 = we + (size_t)m0 * D_DIM + half * 8;
  const bf16_t* bptr1 = we + (size_t)m1 * D_DIM + half * 8;

  v8f c00 = {}, c01 = {}, c10 = {}, c11 = {};
#pragma unroll 2
  for (int ks = 0; ks < D_DIM / 32; ++ks) {
    const int k0 = ks * 32;
    __builtin_prefetch(bptr0 + k0 + 128, 0, 1);
    __builtin_prefetch(bptr1 + k0 + 128, 0, 1);
    v16bf a0 = ld2x16(aptr0 + k0, aptr0 + k0 + 16);
    v16bf a1 = ld2x16(aptr1 + k0, aptr1 + k0 + 16);
    v16bf bb0 = ld2x16(bptr0 + k0, bptr0 + k0 + 16);
    v16bf bb1 = ld2x16(bptr1 + k0, bptr1 + k0 + 16);
    c00 = __builtin_amdgcn_wmma_f32_16x16x32_bf16(false, a0, false, bb0,
                                                  (short)0, c00, false, false);
    c01 = __builtin_amdgcn_wmma_f32_16x16x32_bf16(false, a0, false, bb1,
                                                  (short)0, c01, false, false);
    c10 = __builtin_amdgcn_wmma_f32_16x16x32_bf16(false, a1, false, bb0,
                                                  (short)0, c10, false, false);
    c11 = __builtin_amdgcn_wmma_f32_16x16x32_bf16(false, a1, false, bb1,
                                                  (short)0, c11, false, false);
  }

  const float bias0 = nb[m0];
  const float bias1 = nb[m1];
  const int brow0 = b0 + half * 8;        // b-subtile 0
  const int brow1 = brow0 + 16;           // b-subtile 1
#pragma unroll
  for (int i = 0; i < 8; ++i) {
    pre[(size_t)(brow0 + i) * M_DIM + m0] = c00[i] + bias0;
    pre[(size_t)(brow0 + i) * M_DIM + m1] = c01[i] + bias1;
    pre[(size_t)(brow1 + i) * M_DIM + m0] = c10[i] + bias0;
    pre[(size_t)(brow1 + i) * M_DIM + m1] = c11[i] + bias1;
  }
}

// ---------------------------------------------------------------------------
// Kernel 3: per-row exact top-128 (sorted desc) + masked top-64 (aux)
// one WG (256 thr) per row; row cached in dynamic LDS
// ---------------------------------------------------------------------------
__global__ __launch_bounds__(256) void sae_topk(
    const float* __restrict__ pre, const int* __restrict__ steps,
    float* __restrict__ selv, int* __restrict__ seli,
    float* __restrict__ out_tv, float* __restrict__ out_ti,
    float* __restrict__ out_av, float* __restrict__ out_ai) {
  extern __shared__ unsigned char smem_raw[];
  float* s_vals = (float*)smem_raw;                      // 16384 f32
  float* s_tv   = (float*)(smem_raw + 65536);            // 256
  int*   s_ti   = (int*)(smem_raw + 66560);              // 256
  float* s_selv = (float*)(smem_raw + 67584);            // 192
  int*   s_seli = (int*)(smem_raw + 68352);              // 192
  int*   s_b    = (int*)(smem_raw + 69120);              // broadcast

  const int t = threadIdx.x;
  const int row = blockIdx.x;
  const float* rowp = pre + (size_t)row * M_DIM;

  // ---- pass 1: main top-128 over raw pre_act ----
  for (int j = t; j < M_DIM; j += 256) s_vals[j] = rowp[j];
  __syncthreads();
  {
    float lm = NEG_FLT_MAX; int la = t;
    for (int j = t; j < M_DIM; j += 256) {
      float v = s_vals[j];
      if (v > lm) { lm = v; la = j; }
    }
    s_tv[t] = lm; s_ti[t] = la;
  }
  __syncthreads();

  for (int it = 0; it < MK_SEL; ++it) {
    if (t < 32) {
      float bv = s_tv[t]; int bi = s_ti[t];
      for (int s = 32; s < 256; s += 32) {
        float v = s_tv[t + s]; int ix = s_ti[t + s];
        if (v > bv || (v == bv && ix < bi)) { bv = v; bi = ix; }
      }
      for (int off = 16; off > 0; off >>= 1) {
        float ov = __shfl_xor(bv, off, 32);
        int   oi = __shfl_xor(bi, off, 32);
        if (ov > bv || (ov == bv && oi < bi)) { bv = ov; bi = oi; }
      }
      if (t == 0) { s_selv[it] = bv; s_seli[it] = bi; s_b[0] = bi; }
    }
    __syncthreads();
    const int bi = s_b[0];
    if (t == (bi & 255)) {
      s_vals[bi] = NEG_FLT_MAX;
      float lm = NEG_FLT_MAX; int la = t;
      for (int j = t; j < M_DIM; j += 256) {
        float v = s_vals[j];
        if (v > lm) { lm = v; la = j; }
      }
      s_tv[t] = lm; s_ti[t] = la;
    }
    __syncthreads();
  }

  if (t < MK_SEL) {
    float rv = s_selv[t]; rv = rv > 0.0f ? rv : 0.0f;  // relu
    selv[(size_t)row * MK_SEL + t] = rv;
    seli[(size_t)row * MK_SEL + t] = s_seli[t];
    if (t < K_SEL) {
      out_tv[(size_t)row * K_SEL + t] = rv;
      out_ti[(size_t)row * K_SEL + t] = (float)s_seli[t];
    }
  }

  // ---- pass 2: aux top-64 over pre_act * dead_mask ----
  for (int j = t; j < M_DIM; j += 256)
    s_vals[j] = (steps[j] > THRESH) ? rowp[j] : 0.0f;
  __syncthreads();
  {
    float lm = NEG_FLT_MAX; int la = t;
    for (int j = t; j < M_DIM; j += 256) {
      float v = s_vals[j];
      if (v > lm) { lm = v; la = j; }
    }
    s_tv[t] = lm; s_ti[t] = la;
  }
  __syncthreads();

  for (int it = 0; it < AUX_SEL; ++it) {
    if (t < 32) {
      float bv = s_tv[t]; int bi = s_ti[t];
      for (int s = 32; s < 256; s += 32) {
        float v = s_tv[t + s]; int ix = s_ti[t + s];
        if (v > bv || (v == bv && ix < bi)) { bv = v; bi = ix; }
      }
      for (int off = 16; off > 0; off >>= 1) {
        float ov = __shfl_xor(bv, off, 32);
        int   oi = __shfl_xor(bi, off, 32);
        if (ov > bv || (ov == bv && oi < bi)) { bv = ov; bi = oi; }
      }
      if (t == 0) { s_selv[128 + it] = bv; s_seli[128 + it] = bi; s_b[0] = bi; }
    }
    __syncthreads();
    const int bi = s_b[0];
    if (t == (bi & 255)) {
      s_vals[bi] = NEG_FLT_MAX;
      float lm = NEG_FLT_MAX; int la = t;
      for (int j = t; j < M_DIM; j += 256) {
        float v = s_vals[j];
        if (v > lm) { lm = v; la = j; }
      }
      s_tv[t] = lm; s_ti[t] = la;
    }
    __syncthreads();
  }

  if (t < AUX_SEL) {
    float rv = s_selv[128 + t]; rv = rv > 0.0f ? rv : 0.0f;  // relu
    out_av[(size_t)row * AUX_SEL + t] = rv;
    out_ai[(size_t)row * AUX_SEL + t] = (float)s_seli[128 + t];
  }
}

// ---------------------------------------------------------------------------
// Kernel 4: sparse reconstructions. recon = sum_{k<32} v_k * W_dec[:,i_k] + ib
//           multik uses all 128. W_dec (48 MB) stays L2-resident.
// ---------------------------------------------------------------------------
__global__ __launch_bounds__(256) void sae_recon(
    const float* __restrict__ selv, const int* __restrict__ seli,
    const float* __restrict__ W_dec, const float* __restrict__ ib,
    float* __restrict__ out_rec, float* __restrict__ out_mrec) {
  __shared__ float sv[MK_SEL];
  __shared__ int   si[MK_SEL];
  const int t = threadIdx.x;
  const int row = blockIdx.x;
  if (t < MK_SEL) {
    sv[t] = selv[(size_t)row * MK_SEL + t];
    si[t] = seli[(size_t)row * MK_SEL + t];
  }
  __syncthreads();

  for (int d = t; d < D_DIM; d += 256) {
    const float* wrow = W_dec + (size_t)d * M_DIM;
    float acc32 = 0.0f, acc128 = 0.0f;
#pragma unroll 8
    for (int k = 0; k < MK_SEL; ++k) {
      float c = sv[k] * wrow[si[k]];
      acc128 += c;
      if (k < K_SEL) acc32 += c;
    }
    const float bias = ib[d];
    out_rec[(size_t)row * D_DIM + d]  = acc32 + bias;
    out_mrec[(size_t)row * D_DIM + d] = acc128 + bias;
  }
}

// ---------------------------------------------------------------------------
extern "C" void kernel_launch(void* const* d_in, const int* in_sizes, int n_in,
                              void* d_out, int out_size, void* d_ws, size_t ws_size,
                              hipStream_t stream) {
  const float* x      = (const float*)d_in[0];
  const float* W_enc  = (const float*)d_in[1];
  const float* W_dec  = (const float*)d_in[2];
  const float* ibias  = (const float*)d_in[3];
  const float* nbias  = (const float*)d_in[4];
  const int*   steps  = (const int*)d_in[5];

  char* ws = (char*)d_ws;
  bf16_t* xcb  = (bf16_t*)(ws + OFF_XC);
  bf16_t* web  = (bf16_t*)(ws + OFF_WE);
  float*  pre  = (float*)(ws + OFF_PRE);
  float*  selv = (float*)(ws + OFF_SELV);
  int*    seli = (int*)(ws + OFF_SELI);

  float* out      = (float*)d_out;
  float* out_rec  = out;
  float* out_mrec = out_rec + (size_t)B_DIM * D_DIM;
  float* out_tv   = out_mrec + (size_t)B_DIM * D_DIM;
  float* out_ti   = out_tv + (size_t)B_DIM * K_SEL;
  float* out_av   = out_ti + (size_t)B_DIM * K_SEL;
  float* out_ai   = out_av + (size_t)B_DIM * AUX_SEL;

  // 1) bf16 conversion (covers max(NX, NW) elements)
  {
    const size_t NW = (size_t)M_DIM * D_DIM;
    dim3 grid((unsigned)((NW + 255) / 256));
    sae_convert<<<grid, 256, 0, stream>>>(x, W_enc, ibias, xcb, web);
  }
  // 2) WMMA GEMM: grid = (M/256, B/32)
  {
    dim3 grid(M_DIM / 256, B_DIM / 32);
    sae_gemm<<<grid, 256, 0, stream>>>(xcb, web, nbias, pre);
  }
  // 3) per-row top-k selection (dynamic LDS: 69632 bytes)
  sae_topk<<<B_DIM, 256, 69632, stream>>>(pre, steps, selv, seli,
                                          out_tv, out_ti, out_av, out_ai);
  // 4) sparse reconstructions
  sae_recon<<<B_DIM, 256, 0, stream>>>(selv, seli, W_dec, ibias,
                                       out_rec, out_mrec);
}
